// SelfAttention_12506944766474
// MI455X (gfx1250) — matrix operand order repaired and checked by reading, not compiled, not run on hardware
//
#include <hip/hip_runtime.h>

// ---------------------------------------------------------------------------
// Self-attention (BS=4, SEQ=4096, D_EMB=768, D_ATTN=64), causal, fp32 in/out.
// bf16 WMMA (v_wmma_f32_16x16x32_bf16) for all GEMMs, f32 online softmax.
// Fragments are contiguous B128 accesses (A and B both use k = 16h + s:
// one 32B run/lane). x and W pre-converted to bf16 (W transposed),
// V stored transposed so the P.V B-fragment is contiguous. Row-sums of the
// softmax come from an extra WMMA against an all-ones B-fragment instead of
// lane shuffles; only the single diagonal block per query tile pays for
// causal-mask VALU.
// ---------------------------------------------------------------------------

typedef __attribute__((ext_vector_type(16))) __bf16 v16bf;
typedef __attribute__((ext_vector_type(8)))  float  v8f;
typedef __attribute__((ext_vector_type(2)))  __bf16 bf16x2;

#define BATCH  4
#define SEQL   4096
#define DEMB   768
#define DATTN  64
#define NROWS  (BATCH * SEQL)    // 16384 flattened rows
#define ROWTILES (NROWS / 16)    // 1024 16-row tiles
#define WPB    8                 // waves per 256-thread block
#define SM_SCALE 0.125f          // 1/sqrt(64)

union FragBF {
    v16bf    v;
    unsigned u[8];
    uint4    q[2];
};

__device__ __forceinline__ unsigned pack2_bf16(float a, float b) {
    bf16x2 t = { (__bf16)a, (__bf16)b };      // -> v_cvt_pk_bf16_f32
    return __builtin_bit_cast(unsigned, t);
}
__device__ __forceinline__ unsigned short f32_to_bf16(float f) {
    bf16x2 t = { (__bf16)f, (__bf16)0.f };
    return (unsigned short)(__builtin_bit_cast(unsigned, t) & 0xFFFFu);
}
__device__ __forceinline__ v8f v8f_zero() {
    v8f z = {0.f, 0.f, 0.f, 0.f, 0.f, 0.f, 0.f, 0.f};
    return z;
}

// ---------------------------------------------------------------------------
// Kernel 0a: x (f32) -> x16 (bf16), 8 elements / thread, B128 in & out.
// ---------------------------------------------------------------------------
__global__ __launch_bounds__(256) void cvt_x_kernel(
    const float* __restrict__ x, unsigned short* __restrict__ x16)
{
    size_t t = (size_t)blockIdx.x * blockDim.x + threadIdx.x;  // one per 8 elems
    const float4* xf = (const float4*)x;
    float4 f0 = xf[t * 2 + 0];
    float4 f1 = xf[t * 2 + 1];
    uint4 o;
    o.x = pack2_bf16(f0.x, f0.y);
    o.y = pack2_bf16(f0.z, f0.w);
    o.z = pack2_bf16(f1.x, f1.y);
    o.w = pack2_bf16(f1.z, f1.w);
    ((uint4*)x16)[t] = o;
}

// ---------------------------------------------------------------------------
// Kernel 0b: W_{q,k,v} (f32, [768][64]) -> Wt (bf16, [3][64][768]) transposed.
// ---------------------------------------------------------------------------
__global__ __launch_bounds__(256) void cvt_w_kernel(
    const float* __restrict__ Wq, const float* __restrict__ Wk,
    const float* __restrict__ Wv, unsigned short* __restrict__ Wt)
{
    int t = blockIdx.x * blockDim.x + threadIdx.x;   // 0 .. 3*64*96-1
    int m   = t / (DATTN * (DEMB / 8));
    int rem = t % (DATTN * (DEMB / 8));
    int n   = rem / (DEMB / 8);
    int c0  = (rem % (DEMB / 8)) * 8;
    const float* W = (m == 0) ? Wq : (m == 1) ? Wk : Wv;
    uint4 o;
    o.x = pack2_bf16(W[(size_t)(c0 + 0) * DATTN + n], W[(size_t)(c0 + 1) * DATTN + n]);
    o.y = pack2_bf16(W[(size_t)(c0 + 2) * DATTN + n], W[(size_t)(c0 + 3) * DATTN + n]);
    o.z = pack2_bf16(W[(size_t)(c0 + 4) * DATTN + n], W[(size_t)(c0 + 5) * DATTN + n]);
    o.w = pack2_bf16(W[(size_t)(c0 + 6) * DATTN + n], W[(size_t)(c0 + 7) * DATTN + n]);
    ((uint4*)Wt)[((size_t)m * DATTN + n) * (DEMB / 8) + (c0 / 8)] = o;
}

// ---------------------------------------------------------------------------
// Kernel 1: QKV projection, pure B128 + WMMA hot loop, loads grouped ahead of
// the 4 WMMAs so waits can overlap. One wave = one 16x64 tile of Q, K, or V.
// ---------------------------------------------------------------------------
__global__ __launch_bounds__(256) void qkv_proj_kernel(
    const unsigned short* __restrict__ x16,
    const unsigned short* __restrict__ Wt,
    unsigned short* __restrict__ Q,
    unsigned short* __restrict__ K,
    unsigned short* __restrict__ Vt)
{
    const int lane = threadIdx.x & 31;
    const int wid  = threadIdx.x >> 5;
    const int gw   = blockIdx.x * WPB + wid;          // 0 .. 3071
    const int m    = gw % 3;
    const int rt   = gw / 3;

    const int h  = lane >> 4;
    const int ln = lane & 15;

    const unsigned short* xrow = x16 + (size_t)(rt * 16 + ln) * DEMB;
    const unsigned short* wrow = Wt + (size_t)m * DATTN * DEMB;   // [n][768]

    v8f acc[4] = {v8f_zero(), v8f_zero(), v8f_zero(), v8f_zero()};

    for (int k0 = 0; k0 < DEMB; k0 += 32) {
        FragBF a, b[4];
        a.q[0] = *(const uint4*)(xrow + k0 + 16 * h);
        a.q[1] = *(const uint4*)(xrow + k0 + 16 * h + 8);
#pragma unroll
        for (int nt = 0; nt < 4; ++nt) {
            const unsigned short* wr =
                wrow + (size_t)(nt * 16 + ln) * DEMB + k0 + 16 * h;
            b[nt].q[0] = *(const uint4*)(wr);
            b[nt].q[1] = *(const uint4*)(wr + 8);
        }
#pragma unroll
        for (int nt = 0; nt < 4; ++nt)
            acc[nt] = __builtin_amdgcn_wmma_f32_16x16x32_bf16(
                false, a.v, false, b[nt].v, (short)0, acc[nt], false, false);
    }

    // C/D layout: lane l, VGPR r -> (M = r + 8h, N = l%16)
    if (m == 0 || m == 1) {
        unsigned short* out = (m == 0) ? Q : K;
#pragma unroll
        for (int nt = 0; nt < 4; ++nt)
#pragma unroll
            for (int r = 0; r < 8; ++r) {
                int row = rt * 16 + r + 8 * h;
                out[(size_t)row * DATTN + nt * 16 + ln] = f32_to_bf16(acc[nt][r]);
            }
    } else {
        // transposed store: Vt[b][n][seqpos]
        const int bb = rt / (SEQL / 16);
        const int s0 = rt * 16 - bb * SEQL;
        unsigned short* vb = Vt + (size_t)bb * DATTN * SEQL;
#pragma unroll
        for (int nt = 0; nt < 4; ++nt)
#pragma unroll
            for (int r = 0; r < 8; ++r) {
                int sp = s0 + r + 8 * h;
                vb[(size_t)(nt * 16 + ln) * SEQL + sp] = f32_to_bf16(acc[nt][r]);
            }
    }
}

// ---------------------------------------------------------------------------
// One 32-key flash-attention block. MASKED instantiation only for the single
// diagonal block per query tile. Row-sum comes from wmma(P, ones) into the
// osum accumulator (rescaled by alpha like O), not from lane shuffles.
// ---------------------------------------------------------------------------
template <bool MASKED>
__device__ __forceinline__ void attn_block(
    int kbase, int q0, int h, int ln,
    const unsigned short* __restrict__ kb,    // K + batch row base
    const unsigned short* __restrict__ vtb,   // Vt + batch base
    const FragBF* qf, const FragBF& ones,
    v8f* o, v8f& osum, float* mrow, unsigned short* lp)
{
    // ---- scores: two 16x16 tiles, contraction over 64 features ----
    v8f s[2];
#pragma unroll
    for (int t = 0; t < 2; ++t) {
        FragBF kf0, kf1;
        const unsigned short* krow = kb + (size_t)(kbase + t * 16 + ln) * DATTN;
        kf0.q[0] = *(const uint4*)(krow + 16 * h);
        kf0.q[1] = *(const uint4*)(krow + 16 * h + 8);
        kf1.q[0] = *(const uint4*)(krow + 32 + 16 * h);
        kf1.q[1] = *(const uint4*)(krow + 32 + 16 * h + 8);
        v8f z = v8f_zero();
        z = __builtin_amdgcn_wmma_f32_16x16x32_bf16(
            false, qf[0].v, false, kf0.v, (short)0, z, false, false);
        z = __builtin_amdgcn_wmma_f32_16x16x32_bf16(
            false, qf[1].v, false, kf1.v, (short)0, z, false, false);
        s[t] = z;
    }

    // ---- scale (+ causal mask only on the diagonal block) ----
#pragma unroll
    for (int t = 0; t < 2; ++t)
#pragma unroll
        for (int r = 0; r < 8; ++r) {
            float sv = s[t][r] * SM_SCALE;
            if (MASKED) {
                int query = q0 + r + 8 * h;
                if (kbase + t * 16 + ln > query) sv = -1e30f;
            }
            s[t][r] = sv;
        }

    // ---- online softmax: row max via shfl_xor, exp, alpha rescale ----
#pragma unroll
    for (int r = 0; r < 8; ++r) {
        float rmax = fmaxf(s[0][r], s[1][r]);
#pragma unroll
        for (int d = 1; d < 16; d <<= 1)
            rmax = fmaxf(rmax, __shfl_xor(rmax, d, 32));
        float mnew  = fmaxf(mrow[r], rmax);
        float alpha = __expf(mrow[r] - mnew);
        mrow[r] = mnew;
        s[0][r] = __expf(s[0][r] - mnew);
        s[1][r] = __expf(s[1][r] - mnew);
        o[0][r] *= alpha; o[1][r] *= alpha; o[2][r] *= alpha; o[3][r] *= alpha;
        osum[r] *= alpha;
    }

    // ---- P: C-layout -> A-layout via per-wave LDS ----
#pragma unroll
    for (int r = 0; r < 8; ++r) {
        int M = r + 8 * h;
        lp[M * 32 + ln]      = f32_to_bf16(s[0][r]);
        lp[M * 32 + 16 + ln] = f32_to_bf16(s[1][r]);
    }
    asm volatile("s_wait_dscnt 0" ::: "memory");   // DS in-order per wave
    FragBF pf;
    pf.q[0] = *(const uint4*)(lp + ln * 32 + 16 * h);
    pf.q[1] = *(const uint4*)(lp + ln * 32 + 16 * h + 8);

    // ---- row sums: P . ones  (lands in the same C layout as lrow) ----
    osum = __builtin_amdgcn_wmma_f32_16x16x32_bf16(
        false, pf.v, false, ones.v, (short)0, osum, false, false);

    // ---- O += P . V : contiguous Vt B-frags (contraction = 32 keys) ----
#pragma unroll
    for (int nt = 0; nt < 4; ++nt) {
        FragBF vf;
        const unsigned short* vr =
            vtb + (size_t)(nt * 16 + ln) * SEQL + kbase + 16 * h;
        vf.q[0] = *(const uint4*)(vr);
        vf.q[1] = *(const uint4*)(vr + 8);
        o[nt] = __builtin_amdgcn_wmma_f32_16x16x32_bf16(
            false, pf.v, false, vf.v, (short)0, o[nt], false, false);
    }
}

// ---------------------------------------------------------------------------
// Kernel 2: causal flash attention, one wave per 16-query tile.
// ---------------------------------------------------------------------------
__global__ __launch_bounds__(256) void flash_attn_kernel(
    const unsigned short* __restrict__ Q,
    const unsigned short* __restrict__ K,
    const unsigned short* __restrict__ Vt,
    float* __restrict__ out)
{
    __shared__ unsigned short ldsP[WPB][16 * 32];   // 1 KB per wave

    const int lane = threadIdx.x & 31;
    const int wid  = threadIdx.x >> 5;
    const int gw   = blockIdx.x * WPB + wid;        // 0 .. 1023
    const int bb   = gw >> 8;                       // batch
    const int qt   = gw & 255;                      // query tile in batch
    const int q0   = qt * 16;

    const size_t rbase = (size_t)bb * SEQL * DATTN;
    const unsigned short* kb  = K + rbase;
    const unsigned short* vtb = Vt + (size_t)bb * DATTN * SEQL;

    const int h  = lane >> 4;
    const int ln = lane & 15;

    // Q tile as two A-frags (feature halves 0..31, 32..63)
    FragBF qf[2];
    {
        const unsigned short* qrow = Q + rbase + (size_t)(q0 + ln) * DATTN;
#pragma unroll
        for (int half = 0; half < 2; ++half) {
            qf[half].q[0] = *(const uint4*)(qrow + half * 32 + 16 * h);
            qf[half].q[1] = *(const uint4*)(qrow + half * 32 + 16 * h + 8);
        }
    }

    FragBF ones;   // bf16 1.0 in every slot (B-frag for row-sum WMMA)
#pragma unroll
    for (int v = 0; v < 8; ++v) ones.u[v] = 0x3F803F80u;

    v8f o[4] = {v8f_zero(), v8f_zero(), v8f_zero(), v8f_zero()};
    v8f osum = v8f_zero();
    float mrow[8];
#pragma unroll
    for (int r = 0; r < 8; ++r) mrow[r] = -1e30f;

    unsigned short* lp = ldsP[wid];

    // q0/32 fully-unmasked blocks, then exactly one diagonal (masked) block.
    const int nfull = q0 / 32;
    for (int jb = 0; jb < nfull; ++jb) {
        // prefetch next key block (global_prefetch_b8)
        __builtin_prefetch(kb + (size_t)(jb * 32 + 32 + lane) * DATTN, 0, 3);
        __builtin_prefetch(vtb + (size_t)(lane * 2) * SEQL + jb * 32 + 32, 0, 3);
        attn_block<false>(jb * 32, q0, h, ln, kb, vtb, qf, ones, o, osum, mrow, lp);
    }
    attn_block<true>(nfull * 32, q0, h, ln, kb, vtb, qf, ones, o, osum, mrow, lp);

    // ---- epilogue: one reciprocal per row, then scaled stores ----
    float inv[8];
#pragma unroll
    for (int r = 0; r < 8; ++r) inv[r] = 1.0f / osum[r];

    float* obase = out + rbase;
#pragma unroll
    for (int nt = 0; nt < 4; ++nt)
#pragma unroll
        for (int r = 0; r < 8; ++r) {
            int row = q0 + r + 8 * h;
            obase[(size_t)row * DATTN + nt * 16 + ln] = o[nt][r] * inv[r];
        }
}

// ---------------------------------------------------------------------------
extern "C" void kernel_launch(void* const* d_in, const int* in_sizes, int n_in,
                              void* d_out, int out_size, void* d_ws, size_t ws_size,
                              hipStream_t stream) {
    const float* x  = (const float*)d_in[0];
    const float* Wq = (const float*)d_in[1];
    const float* Wk = (const float*)d_in[2];
    const float* Wv = (const float*)d_in[3];
    float* out = (float*)d_out;

    // workspace layout (bf16 elements): x16 24MB | Wt 288KB | Q 2MB | K 2MB | Vt 2MB
    unsigned short* x16 = (unsigned short*)d_ws;
    unsigned short* Wt  = x16 + (size_t)NROWS * DEMB;
    unsigned short* Q   = Wt  + (size_t)3 * DATTN * DEMB;
    unsigned short* Q_K = Q   + (size_t)NROWS * DATTN;
    unsigned short* Vt  = Q_K + (size_t)NROWS * DATTN;

    dim3 blk(256);
    cvt_x_kernel<<<dim3((NROWS * DEMB / 8) / 256), blk, 0, stream>>>(x, x16);
    cvt_w_kernel<<<dim3((3 * DATTN * (DEMB / 8)) / 256), blk, 0, stream>>>(
        Wq, Wk, Wv, Wt);
    qkv_proj_kernel<<<dim3((ROWTILES * 3) / WPB), blk, 0, stream>>>(
        x16, Wt, Q, Q_K, Vt);
    flash_attn_kernel<<<dim3(ROWTILES / WPB), blk, 0, stream>>>(Q, Q_K, Vt, out);
}
